// TITANS_48756468744673
// MI455X (gfx1250) — matrix operand order repaired and checked by reading, not compiled, not run on hardware
//
#include <hip/hip_runtime.h>
#include <hip/hip_bf16.h>

typedef __attribute__((ext_vector_type(16))) _Float16 v16h;
typedef __attribute__((ext_vector_type(8)))  _Float16 v8h;
typedef __attribute__((ext_vector_type(8)))  float    v8f;
typedef __attribute__((ext_vector_type(4)))  float    v4f;

#define D_MODEL 512
#define N_HEAD  8
#define HEAD_D  64
#define D_FF    2048
#define MEMN    128
#define SEQ     2048
#define BATCH   4
#define ROWS    (BATCH * SEQ)   // 8192

__device__ __forceinline__ v8f wmma16(v16h a, v16h b, v8f c) {
    // D = A(16x32 f16) * B(32x16 f16) + C(16x16 f32)
    return __builtin_amdgcn_wmma_f32_16x16x32_f16(
        false, a, false, b, (short)0, c, false, false);
}

// ---------------------------------------------------------------------------
// A fragment: row-major A[M][K]; lane m = lane&15, K pattern {kb..+7, kb+16..+23}
// Raw (f32) load side: 4 independent 16B vector loads -> convert later.
__device__ __forceinline__ void load_a_raw(const float* A, int lda, int row0, int k0,
                                           v4f r[4]) {
    int lane = threadIdx.x & 31;
    const v4f* p = (const v4f*)(A + (size_t)(row0 + (lane & 15)) * lda
                                + k0 + ((lane < 16) ? 0 : 8));
    r[0] = p[0]; r[1] = p[1];   // floats kb .. kb+7
    r[2] = p[4]; r[3] = p[5];   // floats kb+16 .. kb+23
}
__device__ __forceinline__ v16h cvt_a(const v4f r[4]) {
    v16h a;
#pragma unroll
    for (int j = 0; j < 4; ++j) {
        a[j]      = (_Float16)r[0][j];
        a[4 + j]  = (_Float16)r[1][j];
        a[8 + j]  = (_Float16)r[2][j];
        a[12 + j] = (_Float16)r[3][j];
    }
    return a;
}

// ---------------------------------------------------------------------------
// B tile staging: W row-major [N][K]; B[k][n] = W[n][k].
// LDS layout per buffer: [tile t(4)][lane(32)][j(16)] f16 -> each wave reads a
// contiguous 32B v16h per (t,lane). 256 threads stage 4*32*16 = 2048 halves.
// thread th -> t = th>>6, lane = th&31, jh = (th>>5)&1 (8 contiguous K each).
__device__ __forceinline__ void stageB_load(const float* W, int ldw, int n0, int k0,
                                            v4f& x0, v4f& x1) {
    int th   = threadIdx.x;
    int t    = th >> 6;
    int lane = th & 31;
    int jh   = (th >> 5) & 1;
    int n = n0 + t * 16 + (lane & 15);
    int k = k0 + ((lane < 16) ? 0 : 8) + jh * 16;
    const v4f* p = (const v4f*)(W + (size_t)n * ldw + k);
    x0 = p[0]; x1 = p[1];
}
__device__ __forceinline__ void stageB_store(_Float16* ldsbuf, v4f x0, v4f x1) {
    int th   = threadIdx.x;
    int t    = th >> 6;
    int lane = th & 31;
    int jh   = (th >> 5) & 1;
    v8h h;
#pragma unroll
    for (int j = 0; j < 4; ++j) { h[j] = (_Float16)x0[j]; h[4 + j] = (_Float16)x1[j]; }
    *(v8h*)&ldsbuf[(t * 32 + lane) * 16 + jh * 8] = h;
}

// ---- generic WMMA GEMM: C = act(A[M,K] @ W[N,K]^T + bias) (+ residual) -----
// block = 256 thr = 8 waves; wave -> 16(M) x 64(N); block tile 128M x 64N.
// Requires M%128==0, N%64==0, K%32==0 (true for every call site here).
__global__ __launch_bounds__(256)
void gemm_wmma_kernel(const float* __restrict__ A, int lda,
                      const float* __restrict__ W, int ldw,
                      const float* __restrict__ bias,
                      const float* __restrict__ resid,
                      float* __restrict__ C, int ldc,
                      int M, int N, int K, int act) {
    __shared__ _Float16 ldsB[2][4 * 32 * 16];   // 8 KB, double buffered

    int wid  = threadIdx.x >> 5;
    int row0 = blockIdx.y * 128 + wid * 16;
    int n0   = blockIdx.x * 64;
    int lane = threadIdx.x & 31;

    v8f z = {0.f, 0.f, 0.f, 0.f, 0.f, 0.f, 0.f, 0.f};
    v8f acc[4] = {z, z, z, z};

    v4f araw[4];
    v4f bs0, bs1;
    load_a_raw(A, lda, row0, 0, araw);
    stageB_load(W, ldw, n0, 0, bs0, bs1);
    stageB_store(ldsB[0], bs0, bs1);
    __syncthreads();

    int buf = 0;
    for (int k0 = 0; k0 < K; k0 += 32) {
        v16h a = cvt_a(araw);
        v16h b[4];
#pragma unroll
        for (int t = 0; t < 4; ++t)
            b[t] = *(const v16h*)&ldsB[buf][(t * 32 + lane) * 16];

        bool more = (k0 + 32) < K;
        if (more) {  // issue next-tile global loads before the WMMAs
            load_a_raw(A, lda, row0, k0 + 32, araw);
            stageB_load(W, ldw, n0, k0 + 32, bs0, bs1);
        }
#pragma unroll
        for (int t = 0; t < 4; ++t) acc[t] = wmma16(a, b[t], acc[t]);
        if (more) stageB_store(ldsB[buf ^ 1], bs0, bs1);
        __syncthreads();
        buf ^= 1;
    }

    int nloc  = lane & 15;
    int rbase = (lane < 16) ? 0 : 8;
#pragma unroll
    for (int t = 0; t < 4; ++t) {
        int n   = n0 + t * 16 + nloc;
        float bv = bias ? bias[n] : 0.f;
#pragma unroll
        for (int r = 0; r < 8; ++r) {
            int m   = row0 + rbase + r;
            float v = acc[t][r] + bv;
            if (resid) v += resid[(size_t)m * ldc + n];
            if (act == 1) v = fmaxf(v, 0.f);
            C[(size_t)m * ldc + n] = v;
        }
    }
}

// ---- flash attention: one wave per (b, h, 16-q-row tile) -------------------
// qkv: [B*S, 1536] with q at col h*64, k at 512+h*64, v at 1024+h*64
__global__ __launch_bounds__(32)
void attn_kernel(const float* __restrict__ qkv, float* __restrict__ ctx) {
    const int lane = threadIdx.x;
    const int q0   = blockIdx.x * 16;
    const int h    = blockIdx.y;
    const int b    = blockIdx.z;
    const size_t rs = 3 * D_MODEL;  // 1536
    const float* base = qkv + (size_t)b * SEQ * rs;

    // Q fragments (resident whole kernel): two d-chunks of 32
    v16h aq[2];
    {
        const float* qrow = base + (size_t)(q0 + (lane & 15)) * rs + h * HEAD_D;
#pragma unroll
        for (int c = 0; c < 2; ++c) {
            int kb = c * 32 + ((lane < 16) ? 0 : 8);
            const v4f* p = (const v4f*)(qrow + kb);
            v4f r0 = p[0], r1 = p[1], r2 = p[4], r3 = p[5];
            v4f rr[4] = {r0, r1, r2, r3};
            aq[c] = cvt_a(rr);
        }
    }

    v8f z = {0.f, 0.f, 0.f, 0.f, 0.f, 0.f, 0.f, 0.f};
    v8f cacc[4] = {z, z, z, z};
    float rowmax[8], rowsum[8];
#pragma unroll
    for (int r = 0; r < 8; ++r) { rowmax[r] = -3.0e38f; rowsum[r] = 0.f; }

    __shared__ _Float16 plds[16][32];
    const int rbase = (lane < 16) ? 0 : 8;
    const int nloc  = lane & 15;

    for (int kv0 = 0; kv0 < SEQ; kv0 += 32) {
        // ---- scores: two 16x16 tiles (kv cols 0..15 / 16..31 of this chunk)
        v8f s[2] = {z, z};
#pragma unroll
        for (int half = 0; half < 2; ++half) {
            const float* krow = base + (size_t)(kv0 + half * 16 + nloc) * rs + D_MODEL + h * HEAD_D;
#pragma unroll
            for (int c = 0; c < 2; ++c) {
                int kh = c * 32 + ((lane < 16) ? 0 : 8);
                const v4f* p = (const v4f*)(krow + kh);
                v4f r0 = p[0], r1 = p[1], r2 = p[4], r3 = p[5];
                v16h bk;
#pragma unroll
                for (int j = 0; j < 4; ++j) {
                    bk[j]      = (_Float16)r0[j];
                    bk[4 + j]  = (_Float16)r1[j];
                    bk[8 + j]  = (_Float16)r2[j];
                    bk[12 + j] = (_Float16)r3[j];
                }
                s[half] = wmma16(aq[c], bk, s[half]);
            }
        }
        // ---- online softmax (row stats per 16-lane half; xor 1/2/4/8 stays in half)
        float mnew[8], scl[8];
#pragma unroll
        for (int r = 0; r < 8; ++r) {
            float a0 = s[0][r] * 0.125f, a1 = s[1][r] * 0.125f;
            s[0][r] = a0; s[1][r] = a1;
            float mx = fmaxf(a0, a1);
#pragma unroll
            for (int off = 8; off >= 1; off >>= 1) mx = fmaxf(mx, __shfl_xor(mx, off, 32));
            mnew[r] = fmaxf(rowmax[r], mx);
            scl[r]  = __expf(rowmax[r] - mnew[r]);
            rowmax[r] = mnew[r];
        }
#pragma unroll
        for (int r = 0; r < 8; ++r) {
            float p0 = __expf(s[0][r] - mnew[r]);
            float p1 = __expf(s[1][r] - mnew[r]);
            s[0][r] = p0; s[1][r] = p1;
            float ps = p0 + p1;
#pragma unroll
            for (int off = 8; off >= 1; off >>= 1) ps += __shfl_xor(ps, off, 32);
            rowsum[r] = rowsum[r] * scl[r] + ps;
        }
#pragma unroll
        for (int t = 0; t < 4; ++t)
#pragma unroll
            for (int r = 0; r < 8; ++r) cacc[t][r] *= scl[r];

        // ---- transpose P (C-layout -> A-layout) through LDS, f16
#pragma unroll
        for (int r = 0; r < 8; ++r) {
            plds[rbase + r][nloc]      = (_Float16)s[0][r];
            plds[rbase + r][16 + nloc] = (_Float16)s[1][r];
        }
        __syncthreads();
        v16h ap;
        {
            int m  = lane & 15;
            int kb = (lane < 16) ? 0 : 16;
#pragma unroll
            for (int j = 0; j < 8; ++j) ap[j]     = plds[m][kb + j];
#pragma unroll
            for (int j = 0; j < 8; ++j) ap[8 + j] = plds[m][kb + 8 + j];
        }
        // ---- ctx += P @ V  (contraction over 32 kv, 4 d-tiles)
#pragma unroll
        for (int t = 0; t < 4; ++t) {
            int d  = t * 16 + nloc;
            int kh = (lane < 16) ? 0 : 16;
            float tv[16];
#pragma unroll
            for (int j = 0; j < 16; ++j)
                tv[j] = base[(size_t)(kv0 + kh + j) * rs + 2 * D_MODEL + h * HEAD_D + d];
            v16h bv;
#pragma unroll
            for (int j = 0; j < 16; ++j) bv[j] = (_Float16)tv[j];
            cacc[t] = wmma16(ap, bv, cacc[t]);
        }
        __syncthreads();
    }
    // ---- epilogue: normalize and scatter into ctx[B*S, 512]
#pragma unroll
    for (int t = 0; t < 4; ++t)
#pragma unroll
        for (int r = 0; r < 8; ++r) {
            size_t m = (size_t)b * SEQ + q0 + rbase + r;
            ctx[m * D_MODEL + h * HEAD_D + t * 16 + nloc] = cacc[t][r] / rowsum[r];
        }
}

// ---- residual+LN handled as: GEMM stores (resid added), then row LN --------
__global__ __launch_bounds__(256)
void ln_kernel(float* __restrict__ x, const float* __restrict__ g,
               const float* __restrict__ bta, int M) {
    int wid  = threadIdx.x >> 5;
    int lane = threadIdx.x & 31;
    int row  = blockIdx.x * 8 + wid;
    if (row >= M) return;
    float* p = x + (size_t)row * D_MODEL;
    float vals[16];
    float sum = 0.f;
#pragma unroll
    for (int i = 0; i < 16; ++i) { vals[i] = p[lane + i * 32]; sum += vals[i]; }
#pragma unroll
    for (int off = 16; off >= 1; off >>= 1) sum += __shfl_xor(sum, off, 32);
    float mu = sum * (1.0f / D_MODEL);
    float var = 0.f;
#pragma unroll
    for (int i = 0; i < 16; ++i) { float d = vals[i] - mu; var += d * d; }
#pragma unroll
    for (int off = 16; off >= 1; off >>= 1) var += __shfl_xor(var, off, 32);
    float rsv = rsqrtf(var * (1.0f / D_MODEL) + 1e-5f);
#pragma unroll
    for (int i = 0; i < 16; ++i) {
        int c = lane + i * 32;
        p[c] = (vals[i] - mu) * rsv * g[c] + bta[c];
    }
}

// ---- sequential neural-memory scan: state register-resident ----------------
// block b: 256 threads, thread -> (row = t>>1, half = t&1, 64 cols of state)
__global__ __launch_bounds__(256)
void scan_kernel(const float* __restrict__ kmat, const float* __restrict__ vmat,
                 const float* __restrict__ qmat, float* __restrict__ outs) {
    int b    = blockIdx.x;
    int t    = threadIdx.x;
    int row  = t >> 1;
    int half = t & 1;
    int c0   = half * 64;
    float st[64];
#pragma unroll
    for (int j = 0; j < 64; ++j) st[j] = 0.f;
    __shared__ float sk[MEMN], sv[MEMN], sq[MEMN];

    for (int step = 0; step < SEQ; ++step) {
        size_t roff = ((size_t)b * SEQ + step) * MEMN;
        if (t < MEMN) { sk[t] = kmat[roff + t]; sv[t] = vmat[roff + t]; sq[t] = qmat[roff + t]; }
        __syncthreads();
        // mem_out uses state BEFORE the update
        float part = 0.f;
#pragma unroll
        for (int j = 0; j < 64; ++j) part += st[j] * sq[c0 + j];
        part += __shfl_xor(part, 1, 32);
        if (half == 0) outs[roff + row] = part;
        float vr = sv[row];
#pragma unroll
        for (int j = 0; j < 64; ++j) st[j] = 0.99f * st[j] + 0.01f * vr * sk[c0 + j];
        __syncthreads();
    }
}

// ---- gate (sigmoid) + fuse + decode, last token only -----------------------
__global__ __launch_bounds__(128)
void final_kernel(const float* __restrict__ G, const float* __restrict__ stm,
                  const float* __restrict__ ltm, const float* __restrict__ dec_w,
                  const float* __restrict__ dec_b, float* __restrict__ out) {
    int b = blockIdx.x;
    size_t roff = ((size_t)b * SEQ + (SEQ - 1)) * D_MODEL;
    int t = threadIdx.x;
    float acc = 0.f;
#pragma unroll
    for (int i = 0; i < 4; ++i) {
        int c = t + i * 128;
        float gate  = 1.f / (1.f + __expf(-G[roff + c]));
        float fused = gate * stm[roff + c] + (1.f - gate) * ltm[roff + c];
        acc += fused * dec_w[c];
    }
    __shared__ float red[128];
    red[t] = acc;
    __syncthreads();
    for (int s = 64; s >= 1; s >>= 1) {
        if (t < s) red[t] += red[t + s];
        __syncthreads();
    }
    if (t == 0) out[b] = red[0] + dec_b[0];
}

// ---------------------------------------------------------------------------
static void launch_gemm(const float* A, int lda, const float* W, int ldw,
                        const float* bias, const float* resid, float* C, int ldc,
                        int M, int N, int K, int act, hipStream_t stream) {
    dim3 grid(N / 64, M / 128);
    gemm_wmma_kernel<<<grid, 256, 0, stream>>>(A, lda, W, ldw, bias, resid, C, ldc, M, N, K, act);
}

extern "C" void kernel_launch(void* const* d_in, const int* in_sizes, int n_in,
                              void* d_out, int out_size, void* d_ws, size_t ws_size,
                              hipStream_t stream) {
    (void)in_sizes; (void)n_in; (void)out_size; (void)ws_size;
    const float* x       = (const float*)d_in[0];
    const float* Wi      = (const float*)d_in[1];
    const float* bi      = (const float*)d_in[2];
    const float* qkv_w   = (const float*)d_in[3];
    const float* qkv_b   = (const float*)d_in[4];
    const float* attn_ow = (const float*)d_in[5];
    const float* attn_ob = (const float*)d_in[6];
    const float* ln1_g   = (const float*)d_in[7];
    const float* ln1_b   = (const float*)d_in[8];
    const float* ln2_g   = (const float*)d_in[9];
    const float* ln2_b   = (const float*)d_in[10];
    const float* ff1_w   = (const float*)d_in[11];
    const float* ff1_b   = (const float*)d_in[12];
    const float* ff2_w   = (const float*)d_in[13];
    const float* ff2_b   = (const float*)d_in[14];
    const float* mem_kw  = (const float*)d_in[15];
    const float* mem_kb  = (const float*)d_in[16];
    const float* mem_vw  = (const float*)d_in[17];
    const float* mem_vb  = (const float*)d_in[18];
    const float* mem_qw  = (const float*)d_in[19];
    const float* mem_qb  = (const float*)d_in[20];
    const float* mem_ow  = (const float*)d_in[21];
    const float* mem_ob  = (const float*)d_in[22];
    const float* gate_w  = (const float*)d_in[23];
    const float* gate_b  = (const float*)d_in[24];
    const float* dec_w   = (const float*)d_in[25];
    const float* dec_b   = (const float*)d_in[26];

    // workspace layout (floats); total = 8192*4096 = 33.55M floats (~134 MB)
    float* ws   = (float*)d_ws;
    float* E    = ws;                              // 8192 x 512 (x_emb, then stm in-place)
    float* R1   = E + (size_t)ROWS * D_MODEL;      // 8192 x 2048 (QKV then FF1)
    float* CTX  = R1 + (size_t)ROWS * D_FF;        // 8192 x 512 (ctx, later gate logits)
    float* KM   = CTX + (size_t)ROWS * D_MODEL;    // 8192 x 128
    float* VM   = KM + (size_t)ROWS * MEMN;
    float* QM   = VM + (size_t)ROWS * MEMN;
    float* OUTS = QM + (size_t)ROWS * MEMN;        // 8192 x 128
    float* LTM  = OUTS + (size_t)ROWS * MEMN;      // 8192 x 512
    float* G    = CTX;                             // reuse after encoder finishes

    // 1) x_emb = x @ Wi^T + bi
    launch_gemm(x, 64, Wi, 64, bi, nullptr, E, D_MODEL, ROWS, D_MODEL, 64, 0, stream);

    // 2) neural-memory projections (consume x_emb before encoder overwrites E)
    launch_gemm(E, D_MODEL, mem_kw, D_MODEL, mem_kb, nullptr, KM, MEMN, ROWS, MEMN, D_MODEL, 0, stream);
    launch_gemm(E, D_MODEL, mem_vw, D_MODEL, mem_vb, nullptr, VM, MEMN, ROWS, MEMN, D_MODEL, 0, stream);
    launch_gemm(E, D_MODEL, mem_qw, D_MODEL, mem_qb, nullptr, QM, MEMN, ROWS, MEMN, D_MODEL, 0, stream);

    // 3) sequential scan (latency-critical; register-resident state)
    scan_kernel<<<BATCH, 256, 0, stream>>>(KM, VM, QM, OUTS);

    // 4) ltm = outs @ mem_ow^T + mem_ob
    launch_gemm(OUTS, MEMN, mem_ow, MEMN, mem_ob, nullptr, LTM, D_MODEL, ROWS, D_MODEL, MEMN, 0, stream);

    // 5) encoder layers, stm lives in E (in-place residual chain)
    for (int l = 0; l < 2; ++l) {
        const float* qw  = qkv_w + (size_t)l * 3 * D_MODEL * D_MODEL;
        const float* qb  = qkv_b + (size_t)l * 3 * D_MODEL;
        const float* ow  = attn_ow + (size_t)l * D_MODEL * D_MODEL;
        const float* ob  = attn_ob + (size_t)l * D_MODEL;
        const float* f1w = ff1_w + (size_t)l * D_FF * D_MODEL;
        const float* f1b = ff1_b + (size_t)l * D_FF;
        const float* f2w = ff2_w + (size_t)l * D_MODEL * D_FF;
        const float* f2b = ff2_b + (size_t)l * D_MODEL;

        launch_gemm(E, D_MODEL, qw, D_MODEL, qb, nullptr, R1, 3 * D_MODEL, ROWS, 3 * D_MODEL, D_MODEL, 0, stream);
        attn_kernel<<<dim3(SEQ / 16, N_HEAD, BATCH), 32, 0, stream>>>(R1, CTX);
        launch_gemm(CTX, D_MODEL, ow, D_MODEL, ob, E, E, D_MODEL, ROWS, D_MODEL, D_MODEL, 0, stream);
        ln_kernel<<<ROWS / 8, 256, 0, stream>>>(E, ln1_g + (size_t)l * D_MODEL, ln1_b + (size_t)l * D_MODEL, ROWS);
        launch_gemm(E, D_MODEL, f1w, D_MODEL, f1b, nullptr, R1, D_FF, ROWS, D_FF, D_MODEL, 1, stream);
        launch_gemm(R1, D_FF, f2w, D_FF, f2b, E, E, D_MODEL, ROWS, D_MODEL, D_FF, 0, stream);
        ln_kernel<<<ROWS / 8, 256, 0, stream>>>(E, ln2_g + (size_t)l * D_MODEL, ln2_b + (size_t)l * D_MODEL, ROWS);
    }

    // 6) gate logits: G = stm @ gate_w[:, :512]^T + gb, then += ltm @ gate_w[:, 512:]^T
    launch_gemm(E, D_MODEL, gate_w, 2 * D_MODEL, gate_b, nullptr, G, D_MODEL, ROWS, D_MODEL, D_MODEL, 0, stream);
    launch_gemm(LTM, D_MODEL, gate_w + D_MODEL, 2 * D_MODEL, nullptr, G, G, D_MODEL, ROWS, D_MODEL, D_MODEL, 0, stream);

    // 7) sigmoid gate + fuse + decode on last token only
    final_kernel<<<BATCH, 128, 0, stream>>>(G, E, LTM, dec_w, dec_b, (float*)d_out);
}
